// MultiHeadAttention_35940286333032
// MI455X (gfx1250) — compile-verified, matrix operand order checked by reading, not run on hardware
//
#include <hip/hip_runtime.h>
#include <hip/hip_bf16.h>

#define B_   4
#define S_   2048
#define D_   256
#define H_   8
#define HD_  32
#define SCALE_ 0.0625f   // 1/sqrt(256)

typedef __attribute__((ext_vector_type(16))) __bf16 v16bf;
typedef __attribute__((ext_vector_type(8)))  float  v8f;
typedef __attribute__((ext_vector_type(4)))  unsigned int v4u;
typedef __attribute__((ext_vector_type(4)))  int  v4i;
typedef __attribute__((ext_vector_type(8)))  int  v8i;

#if defined(__has_builtin)
#  if __has_builtin(__builtin_amdgcn_tensor_load_to_lds)
#    define USE_TDM 1
#  endif
#endif
#ifndef USE_TDM
#  define USE_TDM 0
#endif

union Frag16 {
  v16bf v;
  __bf16 h[16];
  uint4 q[2];
};

__device__ __forceinline__ v8f wmma_bf16(const Frag16& a, const Frag16& b, v8f c) {
  return __builtin_amdgcn_wmma_f32_16x16x32_bf16(
      /*neg_a=*/false, a.v, /*neg_b=*/false, b.v,
      /*c_mod=*/(short)0, c, /*reuse_a=*/false, /*reuse_b=*/false);
}

#if USE_TDM
// Issue a TDM 2D tile load: tile (tile0 x tile1) elements of a 2-byte tensor
// (tensor_dim0 x tensor_dim1, row stride stride0) starting at gptr -> LDS at
// lds_off, with LDS padding of 4 DWORDs every 16 DWORDs (row stride 80B).
__device__ __forceinline__ void tdm_load_2d(unsigned lds_off, const void* gptr,
                                            unsigned tdim0, unsigned tdim1,
                                            unsigned stride0,
                                            unsigned tile0, unsigned tile1) {
  const unsigned long long ga = (unsigned long long)(size_t)gptr;
  v4u g0;
  g0[0] = 1u;                                          // count=1, user mode
  g0[1] = lds_off;                                     // lds_addr
  g0[2] = (unsigned)ga;                                // global_addr[31:0]
  g0[3] = (unsigned)((ga >> 32) & 0x01FFFFFFu) | (2u << 30);  // [56:32]|type=2
  v8i g1;
  g1[0] = (int)(0x00010000u                            // data_size = 2B
                | (1u << 20)                           // pad_enable
                | (3u << 22)                           // pad_interval: 16 DW
                | (3u << 25));                         // pad_amount: 4 DW
  g1[1] = (int)(tdim0 << 16);                          // tensor_dim0[15:0]
  g1[2] = (int)((tdim0 >> 16) | (tdim1 << 16));        // dim0 hi | dim1 lo
  g1[3] = (int)((tdim1 >> 16) | (tile0 << 16));        // dim1 hi | tile_dim0
  g1[4] = (int)tile1;                                  // tile_dim1, tile_dim2=0
  g1[5] = (int)stride0;                                // tensor_dim0_stride lo
  g1[6] = 0;                                           // stride0 hi | stride1 lo
  g1[7] = 0;                                           // stride1 hi
  const v4i z4 = {0, 0, 0, 0};
#if __clang_major__ >= 23
  const v8i z8 = {0, 0, 0, 0, 0, 0, 0, 0};
  __builtin_amdgcn_tensor_load_to_lds(g0, g1, z4, z4, z8, 0);
#else
  __builtin_amdgcn_tensor_load_to_lds(g0, g1, z4, z4, 0);
#endif
}
#endif  // USE_TDM

// ---------------------------------------------------------------------------
// Kernel 0: one-shot weight prep.  W[k][n] fp32 -> Wt[n][k] bf16 (transposed
// so B-fragments are contiguous-in-k b128 loads, row n per lane).
// ---------------------------------------------------------------------------
__global__ __launch_bounds__(256) void wcvt_kernel(
    const float* __restrict__ Wq, const float* __restrict__ Wk,
    const float* __restrict__ Wv, const float* __restrict__ Wo,
    __bf16* __restrict__ Wt)      // [4][n=256][k=256]
{
  const int which = blockIdx.y;
  const float* W = (which == 0) ? Wq : (which == 1) ? Wk : (which == 2) ? Wv : Wo;
  __bf16* dst = Wt + (size_t)which * D_ * D_;
  const int idx = blockIdx.x * 256 + threadIdx.x;   // grid.x = 256 -> 65536
  const int k = idx >> 8;
  const int n = idx & 255;
  dst[(size_t)n * D_ + k] = (__bf16)W[idx];         // hw bf16 cvt
}

// ---------------------------------------------------------------------------
// Kernel 1: QKV projections.  out = in @ W + bias -> bf16.
// Q,K stored [b,h,s,hd]; V stored transposed [b,h,hd,s].
// WG tile 64x64, 8 waves x (16x32 via 2 WMMAs per k-step). No LDS: A comes
// from global fp32 (hw cvt), B comes from pre-transposed bf16 weights (L2-hot).
// ---------------------------------------------------------------------------
__global__ __launch_bounds__(256) void proj_kernel(
    const float* __restrict__ qin, const float* __restrict__ kin,
    const float* __restrict__ vin,
    const __bf16* __restrict__ Wt,                  // [3][n][k] bf16
    const float* __restrict__ bq, const float* __restrict__ bk,
    const float* __restrict__ bv,
    __bf16* __restrict__ Qh, __bf16* __restrict__ Kh,
    __bf16* __restrict__ Vt)
{
  const int which = blockIdx.z;                     // 0=Q 1=K 2=V
  const float* A    = (which == 0) ? qin : (which == 1) ? kin : vin;
  const float* bias = (which == 0) ? bq  : (which == 1) ? bk  : bv;
  const __bf16* Wm  = Wt + (size_t)which * D_ * D_;

  const int m0  = blockIdx.x * 64;
  const int n0  = blockIdx.y * 64;
  const int tid = threadIdx.x;
  const int wave = tid >> 5;
  const int lane = tid & 31;
  const int hl = lane & 15;
  const int kb = (lane >> 4) * 8;
  const int wm = (wave & 3) * 16;
  const int wn = (wave >> 2) * 32;

  const int n_g0 = n0 + wn + hl;
  const int n_g1 = n_g0 + 16;
  const __bf16* brow0 = Wm + (size_t)n_g0 * D_;
  const __bf16* brow1 = Wm + (size_t)n_g1 * D_;

  v8f acc0 = {0.f,0.f,0.f,0.f,0.f,0.f,0.f,0.f};
  v8f acc1 = acc0;

  for (int k0 = 0; k0 < D_; k0 += 32) {
    // ---- A fragment: fp32 global -> bf16 (hardware cvt) ----
    Frag16 af;
    {
      const float* ap = A + (size_t)(m0 + wm + hl) * D_ + k0;
      if (k0 + 32 < D_) __builtin_prefetch(ap + 32, 0, 3);  // next A tile
      float4 x0 = *(const float4*)(ap + kb);
      float4 x1 = *(const float4*)(ap + kb + 4);
      float4 x2 = *(const float4*)(ap + 16 + kb);
      float4 x3 = *(const float4*)(ap + 16 + kb + 4);
      af.h[0]=(__bf16)x0.x;  af.h[1]=(__bf16)x0.y;  af.h[2]=(__bf16)x0.z;  af.h[3]=(__bf16)x0.w;
      af.h[4]=(__bf16)x1.x;  af.h[5]=(__bf16)x1.y;  af.h[6]=(__bf16)x1.z;  af.h[7]=(__bf16)x1.w;
      af.h[8]=(__bf16)x2.x;  af.h[9]=(__bf16)x2.y;  af.h[10]=(__bf16)x2.z; af.h[11]=(__bf16)x2.w;
      af.h[12]=(__bf16)x3.x; af.h[13]=(__bf16)x3.y; af.h[14]=(__bf16)x3.z; af.h[15]=(__bf16)x3.w;
    }
    // ---- B fragments straight from pre-transposed bf16 weights ----
    Frag16 bf0, bf1;
    bf0.q[0] = *(const uint4*)(brow0 + k0 + kb);
    bf0.q[1] = *(const uint4*)(brow0 + k0 + 16 + kb);
    bf1.q[0] = *(const uint4*)(brow1 + k0 + kb);
    bf1.q[1] = *(const uint4*)(brow1 + k0 + 16 + kb);

    acc0 = wmma_bf16(af, bf0, acc0);
    acc1 = wmma_bf16(af, bf1, acc1);
  }

  // ---- epilogue: bias add + bf16 store in attention-friendly layout ----
  const int rb = kb;                   // C-layout: M = rb + v, N = hl
  const float bia0 = bias[n_g0];
  const float bia1 = bias[n_g1];

  if (which < 2) {
    __bf16* dst = (which == 0) ? Qh : Kh;
#pragma unroll
    for (int v = 0; v < 8; ++v) {
      const int m_g = m0 + wm + rb + v;
      const int bb  = m_g >> 11;
      const int ss  = m_g & (S_ - 1);
      const int h0 = n_g0 >> 5, d0 = n_g0 & 31;
      const int h1 = n_g1 >> 5, d1 = n_g1 & 31;
      dst[(((size_t)bb * H_ + h0) * S_ + ss) * HD_ + d0] = (__bf16)(acc0[v] + bia0);
      dst[(((size_t)bb * H_ + h1) * S_ + ss) * HD_ + d1] = (__bf16)(acc1[v] + bia1);
    }
  } else {
    // V transposed: 8 consecutive s for fixed hd -> one packed b128 store each
    const int m_g = m0 + wm + rb;
    const int bb  = m_g >> 11;
    const int ss  = m_g & (S_ - 1);
    alignas(16) __bf16 t0[8], t1[8];
#pragma unroll
    for (int v = 0; v < 8; ++v) {
      t0[v] = (__bf16)(acc0[v] + bia0);
      t1[v] = (__bf16)(acc1[v] + bia1);
    }
    const int h0 = n_g0 >> 5, d0 = n_g0 & 31;
    const int h1 = n_g1 >> 5, d1 = n_g1 & 31;
    *(uint4*)&Vt[(((size_t)bb * H_ + h0) * HD_ + d0) * S_ + ss] = *(const uint4*)t0;
    *(uint4*)&Vt[(((size_t)bb * H_ + h1) * HD_ + d1) * S_ + ss] = *(const uint4*)t1;
  }
}

// ---------------------------------------------------------------------------
// Kernel 2: causal attention, block-cooperative K/V streaming with a
// DOUBLE-BUFFERED Tensor Data Mover pipeline.
// Per block: 128 q-rows (8 waves x 16). Wave 0 issues the next chunk's K/V
// tile DMAs into the alternate LDS buffer before waiting on the current one
// (s_wait_tensorcnt <= 2: TDM ops of a wave complete in order, so the older
// pair is guaranteed landed while the new pair is still in flight).
// QK^T: one WMMA per 16-key half; p = mask ? exp(s*SCALE) : 0 (faithful to the
// reference); P bounced through per-wave LDS tile (C->A layout transpose).
// ---------------------------------------------------------------------------
__global__ __launch_bounds__(256) void attn_kernel(
    const __bf16* __restrict__ Qh, const __bf16* __restrict__ Kh,
    const __bf16* __restrict__ Vt, __bf16* __restrict__ Ow)
{
  const int b = blockIdx.z, h = blockIdx.y;
  const int tid  = threadIdx.x;
  const int wave = tid >> 5;
  const int lane = tid & 31;
  const int hl = lane & 15;
  const int kb = (lane >> 4) * 8;
  const int rb = kb;

  const int qblk = blockIdx.x * 128;
  const int q0 = qblk + wave * 16;
  const int bh = b * H_ + h;

  __shared__ alignas(16) __bf16 Kbuf[2][32][40];  // [buf][key][hd], padded
  __shared__ alignas(16) __bf16 Vbuf[2][32][40];  // [buf][hd][key], padded
  __shared__ alignas(16) __bf16 P[8][16][40];     // per-wave P tile

  const __bf16* Qp = Qh + (size_t)bh * S_ * HD_;
  const __bf16* Kp = Kh + (size_t)bh * S_ * HD_;
  const __bf16* Vp = Vt + (size_t)bh * HD_ * S_;

  Frag16 qa;
  {
    const __bf16* p = Qp + (size_t)(q0 + hl) * HD_;
    qa.q[0] = *(const uint4*)(p + kb);
    qa.q[1] = *(const uint4*)(p + 16 + kb);
  }

  v8f acc0 = {0.f,0.f,0.f,0.f,0.f,0.f,0.f,0.f};   // output dims 0..15
  v8f acc1 = acc0;                                 // output dims 16..31
  float lsum[8] = {0.f,0.f,0.f,0.f,0.f,0.f,0.f,0.f};

  const int my_end   = q0 + 15;                    // wave causal limit
  const int nchunks  = (qblk + 128) / 32;          // block-uniform chunk count

#if USE_TDM
  if (wave == 0) {                                 // prologue: fill buffer 0
    tdm_load_2d((unsigned)(size_t)&Kbuf[0][0][0], Kp, HD_, S_, HD_, 32, 32);
    tdm_load_2d((unsigned)(size_t)&Vbuf[0][0][0], Vp, S_, HD_, S_, 32, 32);
  }
#endif

  for (int c = 0; c < nchunks; ++c) {
    const int kv  = c * 32;
    const int cur = c & 1;
#if USE_TDM
    if (wave == 0) {
      if (c + 1 < nchunks) {
        const int nxt = cur ^ 1;
        // issue next chunk into the alternate buffer, then wait only for the
        // current pair (in-order completion => <=2 outstanding is enough)
        tdm_load_2d((unsigned)(size_t)&Kbuf[nxt][0][0],
                    Kp + (size_t)(kv + 32) * HD_, HD_, S_, HD_, 32, 32);
        tdm_load_2d((unsigned)(size_t)&Vbuf[nxt][0][0],
                    Vp + kv + 32, S_, HD_, S_, 32, 32);
        __builtin_amdgcn_s_wait_tensorcnt(2);
      } else {
        __builtin_amdgcn_s_wait_tensorcnt(0);
      }
    }
#else
    // fallback: cooperative staging with plain loads (16B per thread)
    {
      const int half = tid >> 7;                   // 0: K, 1: V
      const int row  = (tid & 127) >> 2;
      const int quad = (tid & 3) * 8;
      if (half == 0) {
        *(uint4*)&Kbuf[cur][row][quad] =
            *(const uint4*)(Kp + (size_t)(kv + row) * HD_ + quad);
      } else {
        *(uint4*)&Vbuf[cur][row][quad] =
            *(const uint4*)(Vp + (size_t)row * S_ + kv + quad);
      }
    }
#endif
    __syncthreads();

    if (kv <= my_end) {
      // ---- K fragments (two 16-key halves) from LDS ----
      Frag16 kf0, kf1;
      {
        const __bf16* p0 = &Kbuf[cur][hl][0];
        kf0.q[0] = *(const uint4*)(p0 + kb);
        kf0.q[1] = *(const uint4*)(p0 + 16 + kb);
        const __bf16* p1 = &Kbuf[cur][16 + hl][0];
        kf1.q[0] = *(const uint4*)(p1 + kb);
        kf1.q[1] = *(const uint4*)(p1 + 16 + kb);
      }
      v8f s0 = {0.f,0.f,0.f,0.f,0.f,0.f,0.f,0.f};
      v8f s1 = s0;
      s0 = wmma_bf16(qa, kf0, s0);
      s1 = wmma_bf16(qa, kf1, s1);

      // ---- scale + causal mask + exp; stash P; row partial sums ----
#pragma unroll
      for (int v = 0; v < 8; ++v) {
        const int qrow = q0 + rb + v;
        const float p0 = (kv + hl      <= qrow) ? __expf(s0[v] * SCALE_) : 0.f;
        const float p1 = (kv + 16 + hl <= qrow) ? __expf(s1[v] * SCALE_) : 0.f;
        lsum[v] += p0 + p1;
        P[wave][rb + v][hl]      = (__bf16)p0;
        P[wave][rb + v][16 + hl] = (__bf16)p1;
      }

      // ---- P as A fragment (LDS round-trip = layout transpose) ----
      Frag16 pa;
      {
        const __bf16* pp = &P[wave][hl][0];
        pa.q[0] = *(const uint4*)(pp + kb);
        pa.q[1] = *(const uint4*)(pp + 16 + kb);
      }
      // ---- V fragments from LDS (rows = dims, contiguous keys) ----
      Frag16 vf0, vf1;
      {
        const __bf16* p0 = &Vbuf[cur][hl][0];
        vf0.q[0] = *(const uint4*)(p0 + kb);
        vf0.q[1] = *(const uint4*)(p0 + 16 + kb);
        const __bf16* p1 = &Vbuf[cur][16 + hl][0];
        vf1.q[0] = *(const uint4*)(p1 + kb);
        vf1.q[1] = *(const uint4*)(p1 + 16 + kb);
      }
      acc0 = wmma_bf16(pa, vf0, acc0);
      acc1 = wmma_bf16(pa, vf1, acc1);
    }
    __syncthreads();
  }

  // ---- finalize: cross-lane row-sum reduce (16-lane halves), normalize ----
#pragma unroll
  for (int v = 0; v < 8; ++v) {
    float ssum = lsum[v];
    ssum += __shfl_xor(ssum, 1, 32);
    ssum += __shfl_xor(ssum, 2, 32);
    ssum += __shfl_xor(ssum, 4, 32);
    ssum += __shfl_xor(ssum, 8, 32);
    const float r = 1.0f / ssum;
    const int srow = q0 + rb + v;
    const size_t base = ((size_t)b * S_ + srow) * D_ + h * HD_;
    Ow[base + hl]      = (__bf16)(acc0[v] * r);
    Ow[base + 16 + hl] = (__bf16)(acc1[v] * r);
  }
}

// ---------------------------------------------------------------------------
// Kernel 3: output projection. out = attn(bf16) @ Wo + bo -> fp32.
// ---------------------------------------------------------------------------
__global__ __launch_bounds__(256) void oproj_kernel(
    const __bf16* __restrict__ Ain, const __bf16* __restrict__ WoT,
    const float* __restrict__ bo, float* __restrict__ out)
{
  const int m0  = blockIdx.x * 64;
  const int n0  = blockIdx.y * 64;
  const int tid = threadIdx.x;
  const int wave = tid >> 5;
  const int lane = tid & 31;
  const int hl = lane & 15;
  const int kb = (lane >> 4) * 8;
  const int wm = (wave & 3) * 16;
  const int wn = (wave >> 2) * 32;

  const int n_g0 = n0 + wn + hl;
  const int n_g1 = n_g0 + 16;
  const __bf16* brow0 = WoT + (size_t)n_g0 * D_;
  const __bf16* brow1 = WoT + (size_t)n_g1 * D_;

  v8f acc0 = {0.f,0.f,0.f,0.f,0.f,0.f,0.f,0.f};
  v8f acc1 = acc0;

  for (int k0 = 0; k0 < D_; k0 += 32) {
    Frag16 af;
    {
      const __bf16* ap = Ain + (size_t)(m0 + wm + hl) * D_ + k0;
      af.q[0] = *(const uint4*)(ap + kb);
      af.q[1] = *(const uint4*)(ap + 16 + kb);
    }
    Frag16 bf0, bf1;
    bf0.q[0] = *(const uint4*)(brow0 + k0 + kb);
    bf0.q[1] = *(const uint4*)(brow0 + k0 + 16 + kb);
    bf1.q[0] = *(const uint4*)(brow1 + k0 + kb);
    bf1.q[1] = *(const uint4*)(brow1 + k0 + 16 + kb);

    acc0 = wmma_bf16(af, bf0, acc0);
    acc1 = wmma_bf16(af, bf1, acc1);
  }

  const int rb = kb;
  const float bia0 = bo[n_g0];
  const float bia1 = bo[n_g1];
#pragma unroll
  for (int v = 0; v < 8; ++v) {
    const size_t m_g = (size_t)(m0 + wm + rb + v);
    out[m_g * D_ + n_g0] = acc0[v] + bia0;
    out[m_g * D_ + n_g1] = acc1[v] + bia1;
  }
}

// ---------------------------------------------------------------------------
extern "C" void kernel_launch(void* const* d_in, const int* in_sizes, int n_in,
                              void* d_out, int out_size, void* d_ws, size_t ws_size,
                              hipStream_t stream) {
  (void)in_sizes; (void)n_in; (void)out_size; (void)ws_size;
  const float* q_in = (const float*)d_in[0];
  const float* k_in = (const float*)d_in[1];
  const float* v_in = (const float*)d_in[2];
  // d_in[3] = causal mask (reconstructed analytically on device)
  const float* Wq = (const float*)d_in[4];
  const float* bq = (const float*)d_in[5];
  const float* Wk = (const float*)d_in[6];
  const float* bk = (const float*)d_in[7];
  const float* Wv = (const float*)d_in[8];
  const float* bv = (const float*)d_in[9];
  const float* Wo = (const float*)d_in[10];
  const float* bo = (const float*)d_in[11];
  float* out = (float*)d_out;

  const size_t per = (size_t)B_ * S_ * D_;     // elements per bf16 buffer
  __bf16* Qh = (__bf16*)d_ws;                  // [B,H,S,HD]
  __bf16* Kh = Qh + per;                       // [B,H,S,HD]
  __bf16* Vt = Kh + per;                       // [B,H,HD,S]
  __bf16* Ow = Vt + per;                       // [B*S, D]
  __bf16* Wt = Ow + per;                       // [4][n=256][k=256] bf16

  dim3 gw(256, 4, 1);
  wcvt_kernel<<<gw, 256, 0, stream>>>(Wq, Wk, Wv, Wo, Wt);

  dim3 gp((B_ * S_) / 64, D_ / 64, 3);
  proj_kernel<<<gp, 256, 0, stream>>>(q_in, k_in, v_in, Wt, bq, bk, bv,
                                      Qh, Kh, Vt);
  dim3 ga(S_ / 128, H_, B_);
  attn_kernel<<<ga, 256, 0, stream>>>(Qh, Kh, Vt, Ow);

  dim3 go((B_ * S_) / 64, D_ / 64, 1);
  oproj_kernel<<<go, 256, 0, stream>>>(Ow, Wt + (size_t)3 * D_ * D_, bo, out);
}
